// MultiHeadedAttention_14096082666249
// MI455X (gfx1250) — matrix operand order repaired and checked by reading, not compiled, not run on hardware
//
#include <hip/hip_runtime.h>
#include <hip/hip_bf16.h>
#include <cstdint>

// ---------------------------------------------------------------------------
// MHA forward for MI455X (gfx1250, wave32, WMMA).
//   out       = (softmax(Q Kt / 8) V) Wo^T + bo        [2,1024,1024] f32
//   new_cache = concat(cache_k|K , cache_v|V)          [2,16,2048,128] f32
// bf16 v_wmma_f32_16x16x32_bf16, f32 accumulate, flash-style online softmax,
// register double-buffered LDS staging, permlane16 VALU row reductions,
// global_prefetch_b8 ahead of the HBM stream.
// ---------------------------------------------------------------------------

typedef __bf16 bf16_t;
typedef __attribute__((ext_vector_type(16))) __bf16 v16bf;
typedef __attribute__((ext_vector_type(8)))  __bf16 v8bf;
typedef __attribute__((ext_vector_type(4)))  __bf16 v4bf;
typedef __attribute__((ext_vector_type(8)))  float  v8f;

#define N_FEAT 1024
#define HEADS  16
#define HDIM   64
#define TQ     1024
#define TK     2048
#define BATCH  2

static __device__ __forceinline__ bf16_t f2bf(float f) {
  return (bf16_t)f;   // native f32->bf16 trunc (RNE) on gfx1250
}

// Build a 16-half operand from two 16-byte chunks (h0..7 from lo, h8..15 from hi).
static __device__ __forceinline__ v16bf ld_op(const bf16_t* lo, const bf16_t* hi) {
  v8bf a = *(const v8bf*)lo;
  v8bf b = *(const v8bf*)hi;
  v16bf r;
#pragma unroll
  for (int i = 0; i < 8; ++i) { r[i] = a[i]; r[i + 8] = b[i]; }
  return r;
}

static __device__ __forceinline__ v8f wmma_bf16(v16bf a, v16bf b, v8f c) {
  return __builtin_amdgcn_wmma_f32_16x16x32_bf16(false, a, false, b, (short)0, c,
                                                 false, false);
}

static __device__ __forceinline__ v8f zero8() {
  v8f z = {0.f, 0.f, 0.f, 0.f, 0.f, 0.f, 0.f, 0.f};
  return z;
}

// VALU butterfly (xor 1/2/4/8 within each 16-lane group) via v_permlane16_b32.
static __device__ __forceinline__ float pl16(float v, int lo, int hi) {
  int s = __builtin_bit_cast(int, v);
  int r = __builtin_amdgcn_permlane16(s, s, lo, hi, false, false);
  return __builtin_bit_cast(float, r);
}
static __device__ __forceinline__ float rmax16(float v) {
  v = fmaxf(v, pl16(v, 0x67452301, 0xEFCDAB89));   // xor 1
  v = fmaxf(v, pl16(v, 0x54761032, 0xDCFE98BA));   // xor 2
  v = fmaxf(v, pl16(v, 0x32107654, 0xBA98FEDC));   // xor 4
  v = fmaxf(v, pl16(v, 0xFEDCBA98, 0x76543210));   // xor 8
  return v;
}
static __device__ __forceinline__ float rsum16(float v) {
  v += pl16(v, 0x67452301, 0xEFCDAB89);
  v += pl16(v, 0x54761032, 0xDCFE98BA);
  v += pl16(v, 0x32107654, 0xBA98FEDC);
  v += pl16(v, 0xFEDCBA98, 0x76543210);
  return v;
}

// ---------------------------------------------------------------------------
// Copy old cache [B,16,1024,128] into first half of new_cache [B,16,2048,128].
// ---------------------------------------------------------------------------
__global__ __launch_bounds__(256)
void cache_copy(const float4* __restrict__ src, float4* __restrict__ dst) {
  const int i  = blockIdx.x * 256 + threadIdx.x;   // 1,048,576 float4 total
  const int bh = i >> 15;                          // / (1024*128/4)
  const int r  = i & 32767;
  dst[(size_t)bh * 65536 + r] = src[i];
}

// ---------------------------------------------------------------------------
// Y[2048,1024] = X[2048,1024] @ W^T + b ; epilogue scatters per `mode`:
//   mode 0: Q -> ws_q bf16 [B,H,Tq,64], scaled by 1/8
//   mode 1: K -> new_cache[b,h,1024+t, 0..63]  f32
//   mode 2: V -> new_cache[b,h,1024+t, 64..127] f32
//   mode 3: out -> d_out f32 [2048,1024]
// Block tile 128x64, 8 waves, each wave 2x2 WMMA C-tiles, K-step 32,
// register double-buffered staging.
// ---------------------------------------------------------------------------
template <bool A_BF16>
__global__ __launch_bounds__(256)
void gemm128x64(const void* __restrict__ Ain, const float* __restrict__ W,
                const float* __restrict__ bias, int mode,
                float* __restrict__ out_f32, bf16_t* __restrict__ out_q,
                float* __restrict__ out_cache) {
  __shared__ __align__(16) bf16_t Alds[128 * 40];   // stride 40 halves = 80B
  __shared__ __align__(16) bf16_t Blds[64 * 40];

  const int tid   = threadIdx.x;
  const int Mbase = blockIdx.y * 128;
  const int Nbase = blockIdx.x * 64;
  const int w = tid >> 5, lane = tid & 31, ln = lane & 15, hf = lane >> 4;
  const int wm = w & 3, wn = w >> 2;
  const int abase = hf ? 16 : 0;    // A-operand contiguous-K base per lane half
  const int bbase = hf ? 16 : 0;    // B-operand contiguous-K base per lane half

  v8f acc[2][2];
#pragma unroll
  for (int i = 0; i < 2; ++i)
#pragma unroll
    for (int j = 0; j < 2; ++j) acc[i][j] = zero8();

  const int lr = tid >> 3;          // 0..31
  const int lc = (tid & 7) * 4;     // 0..28

  float4 arf[4];                    // staged A tile (f32 path)
  v4bf   arb[4];                    // staged A tile (bf16 path)
  float4 brf[2];                    // staged B tile

  auto load_tiles = [&](int k0) {
#pragma unroll
    for (int p = 0; p < 4; ++p) {
      const int grow = Mbase + lr + 32 * p;
      if (A_BF16)
        arb[p] = *(const v4bf*)((const bf16_t*)Ain + (size_t)grow * N_FEAT + k0 + lc);
      else
        arf[p] = *(const float4*)((const float*)Ain + (size_t)grow * N_FEAT + k0 + lc);
    }
#pragma unroll
    for (int p = 0; p < 2; ++p) {
      const int gcol = Nbase + lr + 32 * p;
      brf[p] = *(const float4*)(W + (size_t)gcol * N_FEAT + k0 + lc);
    }
  };
  auto store_tiles = [&]() {
#pragma unroll
    for (int p = 0; p < 4; ++p) {
      bf16_t* dst = &Alds[(lr + 32 * p) * 40 + lc];
      if (A_BF16) {
        *(v4bf*)dst = arb[p];
      } else {
        dst[0] = f2bf(arf[p].x); dst[1] = f2bf(arf[p].y);
        dst[2] = f2bf(arf[p].z); dst[3] = f2bf(arf[p].w);
      }
    }
#pragma unroll
    for (int p = 0; p < 2; ++p) {
      bf16_t* dst = &Blds[(lr + 32 * p) * 40 + lc];
      dst[0] = f2bf(brf[p].x); dst[1] = f2bf(brf[p].y);
      dst[2] = f2bf(brf[p].z); dst[3] = f2bf(brf[p].w);
    }
  };

  load_tiles(0);
  store_tiles();
  __syncthreads();

  for (int k0 = 0; k0 < N_FEAT; k0 += 32) {
    const bool more = (k0 + 32) < N_FEAT;
    if (more) {
      load_tiles(k0 + 32);                          // overlap with WMMA below
      if (k0 + 96 < N_FEAT) {                       // warm GL2 two tiles ahead
        __builtin_prefetch((const char*)Ain +
                           ((size_t)(Mbase + lr) * N_FEAT + k0 + 96) *
                               (A_BF16 ? 2 : 4), 0, 1);
        __builtin_prefetch(W + (size_t)(Nbase + lr) * N_FEAT + k0 + 96, 0, 1);
      }
    }

    v16bf amat[2], bmat[2];
#pragma unroll
    for (int mt = 0; mt < 2; ++mt) {
      const bf16_t* p = &Alds[(wm * 32 + mt * 16 + ln) * 40 + abase];
      amat[mt] = ld_op(p, p + 8);           // 16 contiguous k
    }
#pragma unroll
    for (int nt = 0; nt < 2; ++nt) {
      const bf16_t* p = &Blds[(wn * 32 + nt * 16 + ln) * 40 + bbase];
      bmat[nt] = ld_op(p, p + 8);           // 16 contiguous k
    }
#pragma unroll
    for (int mt = 0; mt < 2; ++mt)
#pragma unroll
      for (int nt = 0; nt < 2; ++nt)
        acc[mt][nt] = wmma_bf16(amat[mt], bmat[nt], acc[mt][nt]);
    __syncthreads();

    if (more) {
      store_tiles();
      __syncthreads();
    }
  }

  // epilogue: C layout -> lane holds (m = r + 8*hf, n = ln)
#pragma unroll
  for (int nt = 0; nt < 2; ++nt) {
    const int col  = Nbase + wn * 32 + nt * 16 + ln;
    const float bv = bias[col];
    const int h = col >> 6, d = col & 63;
#pragma unroll
    for (int mt = 0; mt < 2; ++mt) {
#pragma unroll
      for (int r = 0; r < 8; ++r) {
        const int   row = Mbase + wm * 32 + mt * 16 + r + hf * 8;
        const float v   = acc[mt][nt][r] + bv;
        if (mode == 3) {
          out_f32[(size_t)row * N_FEAT + col] = v;
        } else {
          const int b = row >> 10, tq = row & 1023;
          if (mode == 0) {
            // fold 1/sqrt(64) into Q
            out_q[(((size_t)(b * HEADS + h) * TQ) + tq) * HDIM + d] =
                f2bf(v * 0.125f);
          } else {
            const int off = (mode == 2) ? 64 : 0;
            out_cache[(((size_t)(b * HEADS + h) * TK) + TQ + tq) * 128 + off + d] = v;
          }
        }
      }
    }
  }
}

// ---------------------------------------------------------------------------
// Streaming attention: block = one (b,h) x 128 q rows; 8 waves x 16 rows.
// K/V chunks of 32 keys register-double-buffered into LDS (f32->bf16),
// online softmax with permlane16 VALU reductions, P through per-wave LDS
// bounce guarded by s_wait_dscnt; output x -> ws_x bf16 in [B,Tq,H*D].
// ---------------------------------------------------------------------------
__global__ __launch_bounds__(256)
void attention(const bf16_t* __restrict__ qws, const float* __restrict__ ncache,
               bf16_t* __restrict__ xws) {
  __shared__ __align__(16) bf16_t Klds[32 * 72];    // [key][d], stride 72
  __shared__ __align__(16) bf16_t Vt[64 * 40];      // [d][key], stride 40
  __shared__ __align__(16) bf16_t Pb[8 * 16 * 40];  // per-wave P bounce

  const int tid = threadIdx.x;
  const int w = tid >> 5, lane = tid & 31, ln = lane & 15, hf = lane >> 4;
  const int bh    = blockIdx.y;
  const int qbase = blockIdx.x * 128 + w * 16;
  const int abase = hf ? 16 : 0;
  const int bbase = hf ? 16 : 0;

  // Q operands held in registers for the whole K loop (already /8 scaled)
  const bf16_t* qrow = qws + ((size_t)bh * TQ + qbase + ln) * HDIM;
  const v16bf aq0 = ld_op(qrow + abase,      qrow + abase + 8);        // d 0..31
  const v16bf aq1 = ld_op(qrow + 32 + abase, qrow + 32 + abase + 8);   // d 32..63

  v8f o[4];
#pragma unroll
  for (int j = 0; j < 4; ++j) o[j] = zero8();
  float mrow[8], lrow[8];
#pragma unroll
  for (int r = 0; r < 8; ++r) { mrow[r] = -1e30f; lrow[r] = 0.f; }

  bf16_t* pw = &Pb[w * 16 * 40];

  const int lkey = tid >> 3;        // 0..31
  const int lc   = (tid & 7) * 8;   // 0..56

  float4 kf0, kf1, vf0, vf1;        // staged K/V chunk

  auto load_chunk = [&](int kt) {
    const float* ksrc = ncache + (((size_t)bh * TK) + kt + lkey) * 128 + lc;
    kf0 = *(const float4*)ksrc;
    kf1 = *(const float4*)(ksrc + 4);
    vf0 = *(const float4*)(ksrc + 64);
    vf1 = *(const float4*)(ksrc + 68);
  };
  auto store_chunk = [&]() {
    bf16_t* kd = &Klds[lkey * 72 + lc];
    kd[0] = f2bf(kf0.x); kd[1] = f2bf(kf0.y); kd[2] = f2bf(kf0.z); kd[3] = f2bf(kf0.w);
    kd[4] = f2bf(kf1.x); kd[5] = f2bf(kf1.y); kd[6] = f2bf(kf1.z); kd[7] = f2bf(kf1.w);
    Vt[(lc + 0) * 40 + lkey] = f2bf(vf0.x);
    Vt[(lc + 1) * 40 + lkey] = f2bf(vf0.y);
    Vt[(lc + 2) * 40 + lkey] = f2bf(vf0.z);
    Vt[(lc + 3) * 40 + lkey] = f2bf(vf0.w);
    Vt[(lc + 4) * 40 + lkey] = f2bf(vf1.x);
    Vt[(lc + 5) * 40 + lkey] = f2bf(vf1.y);
    Vt[(lc + 6) * 40 + lkey] = f2bf(vf1.z);
    Vt[(lc + 7) * 40 + lkey] = f2bf(vf1.w);
  };

  load_chunk(0);
  store_chunk();
  __syncthreads();

  for (int kt = 0; kt < TK; kt += 32) {
    const bool more = (kt + 32) < TK;
    if (more) {
      load_chunk(kt + 32);                          // overlap with compute
      if (kt + 64 < TK)                             // warm GL2 two chunks ahead
        __builtin_prefetch(ncache + (((size_t)bh * TK) + kt + 64 + lkey) * 128 + lc,
                           0, 1);
    }

    // S = Q @ K^T : 16 q-rows x 32 keys, K-dim = 64 (two WMMAs per 16-key tile)
    v8f s[2];
#pragma unroll
    for (int j = 0; j < 2; ++j) {
      const bf16_t* kp = &Klds[(j * 16 + ln) * 72 + bbase];
      const v16bf b0 = ld_op(kp,      kp + 8);    // d = bbase + 0..15
      const v16bf b1 = ld_op(kp + 32, kp + 40);   // d = 32 + bbase + 0..15
      v8f c = zero8();
      c    = wmma_bf16(aq0, b0, c);
      s[j] = wmma_bf16(aq1, b1, c);
    }

    // online softmax: each C row lives in one 16-lane group -> permlane16
#pragma unroll
    for (int r = 0; r < 8; ++r) {
      const float s0 = s[0][r], s1 = s[1][r];
      const float rm   = rmax16(fmaxf(s0, s1));
      const float mnew = fmaxf(mrow[r], rm);
      const float corr = __expf(mrow[r] - mnew);
      const float p0   = __expf(s0 - mnew);
      const float p1   = __expf(s1 - mnew);
      const float rs   = rsum16(p0 + p1);
      lrow[r] = lrow[r] * corr + rs;
      mrow[r] = mnew;
#pragma unroll
      for (int j = 0; j < 4; ++j) o[j][r] *= corr;
      const int m = r + hf * 8;                 // C-layout row
      pw[m * 40 + ln]      = f2bf(p0);
      pw[m * 40 + 16 + ln] = f2bf(p1);
    }
    // same-wave LDS RAW: wait DS stores before re-reading as A-operand
    asm volatile("s_wait_dscnt 0" ::: "memory");

    const bf16_t* pp = &pw[ln * 40 + abase];
    const v16bf ap = ld_op(pp, pp + 8);         // P as 16x32 A-operand
#pragma unroll
    for (int j = 0; j < 4; ++j) {
      const bf16_t* vp = &Vt[(j * 16 + ln) * 40 + bbase];
      o[j] = wmma_bf16(ap, ld_op(vp, vp + 8), o[j]);
    }
    __syncthreads();

    if (more) {
      store_chunk();
      __syncthreads();
    }
  }

  // x[b, t, h*64 + d] (bf16) for the output projection
  const int b = bh >> 4, h = bh & 15;
#pragma unroll
  for (int r = 0; r < 8; ++r) {
    const float inv = 1.f / lrow[r];
    bf16_t* xr = xws + (size_t)(b * TQ + qbase + r + hf * 8) * N_FEAT +
                 h * HDIM + ln;
#pragma unroll
    for (int j = 0; j < 4; ++j) xr[j * 16] = f2bf(o[j][r] * inv);
  }
}

// ---------------------------------------------------------------------------
extern "C" void kernel_launch(void* const* d_in, const int* in_sizes, int n_in,
                              void* d_out, int out_size, void* d_ws, size_t ws_size,
                              hipStream_t stream) {
  const float* query = (const float*)d_in[0];
  const float* key   = (const float*)d_in[1];
  const float* value = (const float*)d_in[2];
  const float* cache = (const float*)d_in[3];
  const float* Wq = (const float*)d_in[4];   const float* bq = (const float*)d_in[5];
  const float* Wk = (const float*)d_in[6];   const float* bk = (const float*)d_in[7];
  const float* Wv = (const float*)d_in[8];   const float* bv = (const float*)d_in[9];
  const float* Wo = (const float*)d_in[10];  const float* bo = (const float*)d_in[11];

  float* out    = (float*)d_out;
  float* ncache = out + (size_t)BATCH * TQ * N_FEAT;           // new_cache output

  bf16_t* ws_q = (bf16_t*)d_ws;                                // 4 MB
  bf16_t* ws_x = ws_q + (size_t)BATCH * HEADS * TQ * HDIM;     // 4 MB

  // old cache -> first half of new_cache
  cache_copy<<<4096, 256, 0, stream>>>((const float4*)cache, (float4*)ncache);

  const dim3 g(16, 16);   // N tiles x M tiles (1024/64, 2048/128)
  gemm128x64<false><<<g, 256, 0, stream>>>(query, Wq, bq, 0, nullptr, ws_q, nullptr);
  gemm128x64<false><<<g, 256, 0, stream>>>(key,   Wk, bk, 1, nullptr, nullptr, ncache);
  gemm128x64<false><<<g, 256, 0, stream>>>(value, Wv, bv, 2, nullptr, nullptr, ncache);

  attention<<<dim3(8, 32), 256, 0, stream>>>(ws_q, ncache, ws_x);

  gemm128x64<true><<<g, 256, 0, stream>>>(ws_x, Wo, bo, 3, out, nullptr, nullptr);
}
